// IAGNNModel_36421322670668
// MI455X (gfx1250) — matrix operand order repaired and checked by reading, not compiled, not run on hardware
//
#include <hip/hip_runtime.h>
#include <hip/hip_bf16.h>

typedef __attribute__((ext_vector_type(16))) _Float16 v16h;
typedef __attribute__((ext_vector_type(8)))  float    v8f;

#define HID 128
#define EPSV 1e-5f

// ---------------------------------------------------------------------------
// A-fragment loader: 16x32 f16 A tile per wave, contiguous k order (k = 16h + s).
// lanes 0..15 hold rows M=lane, K = {k0..k0+15}
// lanes 16..31 hold rows M=lane-16, K = {k0+16..k0+31}
// ---------------------------------------------------------------------------
__device__ inline v16h load_a_frag(const float* __restrict__ A, int row0, int k0, int M) {
    int lane = threadIdx.x & 31;
    int m = lane & 15;
    int r = row0 + m;
    if (r >= M) r = M - 1;                     // clamp (stores are guarded)
    int kb = k0 + ((lane >> 4) << 4);          // +16 for lanes 16..31
    const float* p = A + (size_t)r * HID + kb;
    v16h a;
#pragma unroll
    for (int j = 0; j < 8; ++j) a[j] = (_Float16)p[j];          // K = kb..kb+7
#pragma unroll
    for (int j = 0; j < 8; ++j) a[8 + j] = (_Float16)p[8 + j];  // K = kb+8..kb+15
    return a;
}

// Scatter one f32 element of a row-major [128][ncols] B matrix into the
// LDS fragment layout Bf[kt][nt][lane][j].
__device__ inline void fill_b_frag(_Float16* Bf, const float* __restrict__ B, int tid,
                                   int stride /*threads*/) {
    for (int i = tid; i < HID * HID; i += stride) {
        int n = i & 127, k = i >> 7;
        _Float16 v = (_Float16)B[i];
        int kt = k >> 5, kin = k & 31;
        int j = kin & 15, lanehi = kin >> 4;
        int nt = n >> 4, lane = (lanehi << 4) | (n & 15);
        Bf[(((kt * 8 + nt) * 32) + lane) * 16 + j] = v;
    }
}

// ---------------------------------------------------------------------------
// Input GEMM: h = relu(x @ W_in + b_in)
// Block = 256 threads (8 waves), block tile 128x128, wave tile 16x128.
// ---------------------------------------------------------------------------
__global__ void k_gemm(const float* __restrict__ A, const float* __restrict__ B,
                       const float* __restrict__ bias, float* __restrict__ C,
                       int M, int do_relu) {
    __shared__ __align__(32) _Float16 Bf[4 * 8 * 32 * 16];   // 32 KB
    fill_b_frag(Bf, B, threadIdx.x, blockDim.x);
    __syncthreads();

    int wave = threadIdx.x >> 5;
    int lane = threadIdx.x & 31;
    int row0 = blockIdx.x * 128 + wave * 16;

    v8f acc[8] = {};
#pragma unroll
    for (int kt = 0; kt < 4; ++kt) {
        if (kt < 3) {
            int r = row0 + (lane & 15); if (r >= M) r = M - 1;
            __builtin_prefetch(A + (size_t)r * HID + (kt + 1) * 32, 0, 0);
        }
        v16h a = load_a_frag(A, row0, kt * 32, M);
#pragma unroll
        for (int nt = 0; nt < 8; ++nt) {
            v16h b = *(const v16h*)&Bf[(((kt * 8 + nt) * 32) + lane) * 16];
            acc[nt] = __builtin_amdgcn_wmma_f32_16x16x32_f16(
                false, a, false, b, (short)0, acc[nt], false, false);
        }
    }

    int ncol = lane & 15;
    int roff = (lane >> 4) * 8;
#pragma unroll
    for (int nt = 0; nt < 8; ++nt) {
        int col = nt * 16 + ncol;
        float bv = bias[col];
#pragma unroll
        for (int v = 0; v < 8; ++v) {
            int r = row0 + roff + v;
            if (r < M) {
                float x = acc[nt][v] + bv;
                if (do_relu) x = fmaxf(x, 0.f);
                C[(size_t)r * HID + col] = x;
            }
        }
    }
}

// ---------------------------------------------------------------------------
// Fused per-layer node kernel: one pass over h computes
//   m   = h @ Wm + bm
//   pre = h @ Wr + br
//   a_src = h @ Wa[:128],  a_dst = h @ Wa[128:]   (as cols 0/1 of a WMMA tile)
// 17 WMMA accumulators per wave, A fragment reused 17x per load.
// ---------------------------------------------------------------------------
__global__ void k_layer(const float* __restrict__ h,
                        const float* __restrict__ Wm, const float* __restrict__ bm,
                        const float* __restrict__ Wr, const float* __restrict__ br,
                        const float* __restrict__ Wa,
                        float* __restrict__ m, float* __restrict__ pre,
                        float* __restrict__ a_s, float* __restrict__ a_d, int M) {
    __shared__ __align__(32) _Float16 Bm[4 * 8 * 32 * 16];   // 32 KB
    __shared__ __align__(32) _Float16 Br[4 * 8 * 32 * 16];   // 32 KB
    __shared__ __align__(32) _Float16 Ba[4 * 32 * 16];       //  4 KB (one nt tile)

    fill_b_frag(Bm, Wm, threadIdx.x, blockDim.x);
    fill_b_frag(Br, Wr, threadIdx.x, blockDim.x);
    // Wa tile: col0 = src weights, col1 = dst weights, cols 2..15 = 0
    for (int i = threadIdx.x; i < 4 * 32 * 16; i += blockDim.x) {
        int j = i & 15, lane = (i >> 4) & 31, kt = i >> 9;
        int k = kt * 32 + ((lane >> 4) << 4) + j;
        int n = lane & 15;
        float v = (n == 0) ? Wa[k] : (n == 1 ? Wa[HID + k] : 0.f);
        Ba[i] = (_Float16)v;
    }
    __syncthreads();

    int wave = threadIdx.x >> 5;
    int lane = threadIdx.x & 31;
    int row0 = blockIdx.x * 128 + wave * 16;

    v8f accm[8] = {};
    v8f accr[8] = {};
    v8f acca = {};
#pragma unroll
    for (int kt = 0; kt < 4; ++kt) {
        if (kt < 3) {
            int r = row0 + (lane & 15); if (r >= M) r = M - 1;
            __builtin_prefetch(h + (size_t)r * HID + (kt + 1) * 32, 0, 0);
        }
        v16h a = load_a_frag(h, row0, kt * 32, M);
#pragma unroll
        for (int nt = 0; nt < 8; ++nt) {
            v16h b = *(const v16h*)&Bm[(((kt * 8 + nt) * 32) + lane) * 16];
            accm[nt] = __builtin_amdgcn_wmma_f32_16x16x32_f16(
                false, a, false, b, (short)0, accm[nt], false, false);
        }
#pragma unroll
        for (int nt = 0; nt < 8; ++nt) {
            v16h b = *(const v16h*)&Br[(((kt * 8 + nt) * 32) + lane) * 16];
            accr[nt] = __builtin_amdgcn_wmma_f32_16x16x32_f16(
                false, a, false, b, (short)0, accr[nt], false, false);
        }
        {
            v16h b = *(const v16h*)&Ba[((kt * 32) + lane) * 16];
            acca = __builtin_amdgcn_wmma_f32_16x16x32_f16(
                false, a, false, b, (short)0, acca, false, false);
        }
    }

    int ncol = lane & 15;
    int roff = (lane >> 4) * 8;
#pragma unroll
    for (int nt = 0; nt < 8; ++nt) {
        int col = nt * 16 + ncol;
        float bvm = bm[col];
        float bvr = br[col];
#pragma unroll
        for (int v = 0; v < 8; ++v) {
            int r = row0 + roff + v;
            if (r < M) {
                m[(size_t)r * HID + col]   = accm[nt][v] + bvm;
                pre[(size_t)r * HID + col] = accr[nt][v] + bvr;
            }
        }
    }
    if (ncol < 2) {
        float* dst = (ncol == 0) ? a_s : a_d;
#pragma unroll
        for (int v = 0; v < 8; ++v) {
            int r = row0 + roff + v;
            if (r < M) dst[r] = acca[v];
        }
    }
}

// ---------------------------------------------------------------------------
// Edge kernel: wave per edge. gate = sigmoid(a_src[s]+a_dst[d]+ba);
// pre[dst] += gate * m[src]   (f32 global atomics; L2-resident)
// ---------------------------------------------------------------------------
__global__ void k_edge(const int* __restrict__ src, const int* __restrict__ dst,
                       const float* __restrict__ a_src, const float* __restrict__ a_dst,
                       const float* __restrict__ ba_ptr,
                       const float* __restrict__ m, float* __restrict__ pre, int E) {
    int e = (int)((blockIdx.x * blockDim.x + threadIdx.x) >> 5);
    if (e >= E) return;
    int lane = threadIdx.x & 31;
    int s = src[e], d = dst[e];
    float g = a_src[s] + a_dst[d] + ba_ptr[0];
    g = 1.0f / (1.0f + __expf(-g));
    const float4 mv = *(const float4*)(m + (size_t)s * HID + lane * 4);
    float* pp = pre + (size_t)d * HID + lane * 4;
    atomicAdd(pp + 0, g * mv.x);
    atomicAdd(pp + 1, g * mv.y);
    atomicAdd(pp + 2, g * mv.z);
    atomicAdd(pp + 3, g * mv.w);
}

// ---------------------------------------------------------------------------
// Column statistics over N rows (axis=0 mean/var), thread = column.
// ---------------------------------------------------------------------------
__global__ void k_stats(const float* __restrict__ pre, float* __restrict__ sum,
                        float* __restrict__ sumsq, int N, int chunk) {
    int c = threadIdx.x;
    int r0 = blockIdx.x * chunk;
    int r1 = r0 + chunk; if (r1 > N) r1 = N;
    float s = 0.f, s2 = 0.f;
    for (int r = r0; r < r1; ++r) {
        float v = pre[(size_t)r * HID + c];
        s += v; s2 += v * v;
    }
    atomicAdd(&sum[c], s);
    atomicAdd(&sumsq[c], s2);
}

__global__ void k_finalize(const float* __restrict__ sum, const float* __restrict__ sumsq,
                           float* __restrict__ mu, float* __restrict__ rstd, int N) {
    int c = threadIdx.x;
    float inv = 1.0f / (float)N;
    float mv = sum[c] * inv;
    float var = sumsq[c] * inv - mv * mv;
    mu[c] = mv;
    rstd[c] = rsqrtf(var + EPSV);
}

__global__ void k_norm(const float* __restrict__ pre, const float* __restrict__ mu,
                       const float* __restrict__ rstd, const float* __restrict__ gamma,
                       const float* __restrict__ beta, float* __restrict__ hout, int total) {
    int i = blockIdx.x * blockDim.x + threadIdx.x;
    if (i >= total) return;
    int c = i & 127;
    float x = (pre[i] - mu[c]) * rstd[c] * gamma[c] + beta[c];
    hout[i] = fmaxf(x, 0.f);
}

// ---------------------------------------------------------------------------
// Pool: batch sorted -> run-length accumulate, flush atomics on graph change.
// ---------------------------------------------------------------------------
__global__ void k_pool(const float* __restrict__ h, const int* __restrict__ batch,
                       float* __restrict__ pooled, int N, int chunk) {
    int c = threadIdx.x;                     // 128 threads
    int r0 = blockIdx.x * chunk;
    int r1 = r0 + chunk; if (r1 > N) r1 = N;
    if (r0 >= r1) return;
    float acc = 0.f;
    int cur = batch[r0];
    for (int r = r0; r < r1; ++r) {
        int b = batch[r];
        if (b != cur) {
            atomicAdd(&pooled[(size_t)cur * HID + c], acc);
            acc = 0.f; cur = b;
        }
        acc += h[(size_t)r * HID + c];
    }
    atomicAdd(&pooled[(size_t)cur * HID + c], acc);
}

// ---------------------------------------------------------------------------
// Head: hid = relu(pooled@W1+b1); logits = (hid@W2+b2)/TEMP.  One block.
// ---------------------------------------------------------------------------
__global__ void k_head(const float* __restrict__ pooled, const float* __restrict__ W1,
                       const float* __restrict__ b1, const float* __restrict__ W2,
                       const float* __restrict__ b2, float* __restrict__ out, int G) {
    __shared__ float hid[64 * 64];
    for (int idx = threadIdx.x; idx < G * 64; idx += blockDim.x) {
        int g = idx >> 6, j = idx & 63;
        float s = b1[j];
        for (int k = 0; k < HID; ++k) s += pooled[g * HID + k] * W1[k * 64 + j];
        hid[idx] = fmaxf(s, 0.f);
    }
    __syncthreads();
    for (int idx = threadIdx.x; idx < G * 10; idx += blockDim.x) {
        int g = idx / 10, c = idx % 10;
        float s = b2[c];
        for (int k = 0; k < 64; ++k) s += hid[g * 64 + k] * W2[k * 10 + c];
        out[idx] = s * 0.5f;   // / TEMP (2.0)
    }
}

__global__ void k_zero(float* __restrict__ p, int n) {
    int i = blockIdx.x * blockDim.x + threadIdx.x;
    if (i < n) p[i] = 0.f;
}

// ---------------------------------------------------------------------------
extern "C" void kernel_launch(void* const* d_in, const int* in_sizes, int n_in,
                              void* d_out, int out_size, void* d_ws, size_t ws_size,
                              hipStream_t stream) {
    const float* x      = (const float*)d_in[0];
    const int*   edge   = (const int*)  d_in[1];   // [2,E]: src then dst
    const int*   batch  = (const int*)  d_in[2];
    const float* W_in   = (const float*)d_in[3];
    const float* b_in   = (const float*)d_in[4];
    const float* Wa     = (const float*)d_in[5];   // [L, 2H, 1]
    const float* ba     = (const float*)d_in[6];   // [L, 1]
    const float* Wm     = (const float*)d_in[7];   // [L, H, H]
    const float* bm     = (const float*)d_in[8];
    const float* Wr     = (const float*)d_in[9];
    const float* br     = (const float*)d_in[10];
    const float* gamma  = (const float*)d_in[11];
    const float* beta   = (const float*)d_in[12];
    const float* W1     = (const float*)d_in[13];
    const float* b1     = (const float*)d_in[14];
    const float* W2     = (const float*)d_in[15];
    const float* b2     = (const float*)d_in[16];
    float* out = (float*)d_out;

    const int N = in_sizes[2];
    const int E = in_sizes[1] / 2;
    const int G = out_size / 10;
    const int L = 4;

    // workspace carve-up (floats)
    float* ws     = (float*)d_ws;
    size_t nh     = (size_t)N * HID;
    float* h      = ws;
    float* m      = h + nh;
    float* pre    = m + nh;
    float* a_s    = pre + nh;
    float* a_d    = a_s + N;
    float* sum    = a_d + N;
    float* sumsq  = sum + HID;
    float* mu     = sumsq + HID;
    float* rstd   = mu + HID;
    float* pooled = rstd + HID;           // G*HID

    const int MT      = (N + 127) / 128;  // GEMM M-tiles
    const int WPB     = 8;                // waves per 256-thread block
    const int edgeGrd = (E + WPB - 1) / WPB;
    const int statChunk = 256;
    const int statGrd = (N + statChunk - 1) / statChunk;
    const int normGrd = (int)((nh + 255) / 256);
    const int poolChunk = 512;
    const int poolGrd = (N + poolChunk - 1) / poolChunk;

    // h = relu(x @ W_in + b_in)
    k_gemm<<<MT, 256, 0, stream>>>(x, W_in, b_in, h, N, 1);

    for (int l = 0; l < L; ++l) {
        k_layer<<<MT, 256, 0, stream>>>(h,
                                        Wm + (size_t)l * HID * HID, bm + l * HID,
                                        Wr + (size_t)l * HID * HID, br + l * HID,
                                        Wa + (size_t)l * 2 * HID,
                                        m, pre, a_s, a_d, N);
        k_zero<<<1, 256, 0, stream>>>(sum, 2 * HID);  // sum + sumsq contiguous
        k_edge<<<edgeGrd, 256, 0, stream>>>(edge, edge + E, a_s, a_d, ba + l, m, pre, E);
        k_stats<<<statGrd, HID, 0, stream>>>(pre, sum, sumsq, N, statChunk);
        k_finalize<<<1, HID, 0, stream>>>(sum, sumsq, mu, rstd, N);
        k_norm<<<normGrd, 256, 0, stream>>>(pre, mu, rstd,
                                            gamma + l * HID, beta + l * HID, h, (int)nh);
    }

    k_zero<<<(G * HID + 255) / 256, 256, 0, stream>>>(pooled, G * HID);
    k_pool<<<poolGrd, HID, 0, stream>>>(h, batch, pooled, N, poolChunk);
    k_head<<<1, 256, 0, stream>>>(pooled, W1, b1, W2, b2, out, G);
}
